// CrossAttentionEnhanced_47390669144484
// MI455X (gfx1250) — matrix operand order repaired and checked by reading, not compiled, not run on hardware
//
#include <hip/hip_runtime.h>

// ---------------------------------------------------------------------------
// CDNA5 / gfx1250 implementation of CrossAttentionEnhanced
//   wave32, WMMA (v_wmma_f32_16x16x32_bf16 for big GEMMs,
//                 v_wmma_f32_16x16x4_f32 for channel-attention logits),
//   Tensor Data Mover (tensor_load_to_lds) staging of WMMA B-tiles into LDS.
// ---------------------------------------------------------------------------

typedef __attribute__((ext_vector_type(16))) __bf16 bf16x16;
typedef __attribute__((ext_vector_type(8)))  float  f32x8;
typedef __attribute__((ext_vector_type(2)))  float  f32x2;
typedef __attribute__((ext_vector_type(4)))  unsigned u32x4;
typedef __attribute__((ext_vector_type(4)))  unsigned tdm_g0_t;
typedef __attribute__((ext_vector_type(8)))  int      tdm_g1_t;
typedef __attribute__((ext_vector_type(4)))  int      tdm_g23_t;

#define HW    96
#define NPOS  9216           // 96*96
#define DIM   256
#define HEADS 8
#define CINCH 258            // 256 + 2 coord channels
#define KPAD1 288            // 258 padded to multiple of 32
#define GP1   18             // 288/16 channel groups
#define GP2   16             // 256/16 channel groups (out_proj input)

union BfFrag { u32x4 q[2]; bf16x16 v; };

__device__ __forceinline__ unsigned short f2bf(float f) {
    unsigned u = __float_as_uint(f);
    unsigned r = u + 0x7FFFu + ((u >> 16) & 1u);   // round-to-nearest-even
    return (unsigned short)(r >> 16);
}

// ---------------------------------------------------------------------------
// K0: pack an fp32 [M][K] weight into bf16 [M][Kpad] (zero pad K..Kpad)
// ---------------------------------------------------------------------------
__global__ __launch_bounds__(256) void k_pack_w(
    const float* __restrict__ src, unsigned short* __restrict__ dst,
    int M, int K, int Kpad)
{
    int i = blockIdx.x * 256 + threadIdx.x;
    if (i >= M * Kpad) return;
    int m = i / Kpad, k = i - m * Kpad;
    dst[i] = (k < K) ? f2bf(src[m * K + k]) : (unsigned short)0;
}

// ---------------------------------------------------------------------------
// K1: fused add_coords + channel-LayerNorm + depthwise 3x3 conv, emitting the
//     bf16 channel-pair-packed activation layout for WMMA B fragments:
//       dword D[b][g][n][j] = (ch 16g+2j , ch 16g+2j+1) at spatial position n
// ---------------------------------------------------------------------------
__global__ __launch_bounds__(256) void k_ln_dwconv_pack(
    const float* __restrict__ src,     // [B][256][96][96]
    const float* __restrict__ lnw, const float* __restrict__ lnb,   // [258]
    const float* __restrict__ dww,                                   // [258][9]
    unsigned* __restrict__ dst)        // [B][GP1][NPOS][8] dwords
{
    __shared__ float s_mu[324], s_rs[324], s_tile[324];

    const int b  = blockIdx.z;
    const int bx = blockIdx.x, by = blockIdx.y;
    const int tid = threadIdx.x;
    const float* sb = src + (size_t)b * DIM * NPOS;

    // pass 1: LN statistics for every halo position (over 258 channels)
    for (int hp = tid; hp < 324; hp += 256) {
        int hy = hp / 18, hx = hp - hy * 18;
        int y = by * 16 + hy - 1, x = bx * 16 + hx - 1;
        float mu = 0.f, rs = 0.f;
        if (y >= 0 && y < HW && x >= 0 && x < HW) {
            float xs = -1.f + 2.f * (float)x * (1.f / 95.f);
            float ys = -1.f + 2.f * (float)y * (1.f / 95.f);
            float s  = xs + ys;
            float s2 = xs * xs + ys * ys;
            const float* p = sb + y * HW + x;
            for (int c = 0; c < DIM; ++c) {
                float v = p[(size_t)c * NPOS];
                s += v; s2 += v * v;
            }
            mu = s * (1.f / (float)CINCH);
            float var = s2 * (1.f / (float)CINCH) - mu * mu;
            rs = rsqrtf(var + 1e-6f);
        }
        s_mu[hp] = mu; s_rs[hp] = rs;
    }
    __syncthreads();

    const int iy = tid >> 4, ix = tid & 15;
    const int n  = (by * 16 + iy) * HW + (bx * 16 + ix);

    for (int g = 0; g < GP1; ++g) {
        unsigned pk[8];
        for (int c16 = 0; c16 < 16; ++c16) {
            const int c = g * 16 + c16;
            for (int hp = tid; hp < 324; hp += 256) {
                float v = 0.f;
                if (c < CINCH) {
                    int hy = hp / 18, hx = hp - hy * 18;
                    int y = by * 16 + hy - 1, x = bx * 16 + hx - 1;
                    if (y >= 0 && y < HW && x >= 0 && x < HW) {
                        float raw;
                        if (c < DIM)       raw = sb[(size_t)c * NPOS + y * HW + x];
                        else if (c == 256) raw = -1.f + 2.f * (float)x * (1.f / 95.f);
                        else               raw = -1.f + 2.f * (float)y * (1.f / 95.f);
                        v = (raw - s_mu[hp]) * s_rs[hp] * lnw[c] + lnb[c];
                    }
                }
                s_tile[hp] = v;
            }
            __syncthreads();
            float conv = 0.f;
            if (c < CINCH) {
                const float* w = dww + c * 9;
                int t = iy * 18 + ix;
                conv = s_tile[t     ] * w[0] + s_tile[t +  1] * w[1] + s_tile[t +  2] * w[2]
                     + s_tile[t + 18] * w[3] + s_tile[t + 19] * w[4] + s_tile[t + 20] * w[5]
                     + s_tile[t + 36] * w[6] + s_tile[t + 37] * w[7] + s_tile[t + 38] * w[8];
            }
            unsigned short hb = f2bf(conv);
            if ((c16 & 1) == 0) pk[c16 >> 1]  = (unsigned)hb;
            else                pk[c16 >> 1] |= ((unsigned)hb) << 16;
            __syncthreads();
        }
        u32x4* pd = (u32x4*)(dst + (((size_t)b * GP1 + g) * NPOS + n) * 8);
        u32x4 lo = { pk[0], pk[1], pk[2], pk[3] };
        u32x4 hi = { pk[4], pk[5], pk[6], pk[7] };
        pd[0] = lo; pd[1] = hi;
    }
}

// ---------------------------------------------------------------------------
// K2/K7: bf16 WMMA GEMM  C[b][M][NPOS] = W[M][Kpad] * X[b][Kpad][NPOS]
//   block = 128(M) x 128(N), 8 waves (2 M x 4 N), wave = 4x2 WMMA tiles.
//   B tiles (8 dwords x 128 cols x 2 K-groups = 8KB) are DMA'd into LDS by
//   the Tensor Data Mover (double buffered, one issuing wave, TENSORcnt
//   tracked), then consumed with two ds_read_b128 per lane per fragment.
//   A fragments come straight from L2 via global_load_b128.
// ---------------------------------------------------------------------------
__global__ __launch_bounds__(256) void k_gemm_bf16(
    const unsigned short* __restrict__ Wp,  // [M][Kpad] bf16
    const unsigned* __restrict__ Xp,        // [B][Kpad/16][NPOS][8] dwords
    float* __restrict__ C,                  // [B][M][NPOS]
    int M, int Kpad,
    const float* __restrict__ bias,         // [M] or nullptr
    const float* __restrict__ resid)        // [B][M][NPOS] or nullptr
{
    __shared__ unsigned bufB[2][2048];      // 2 x 8KB: [g(2)][n(128)][j(8)]

    const int b    = blockIdx.z;
    const int G    = Kpad >> 4;
    const int tid  = threadIdx.x;
    const int wid  = tid >> 5;
    const int lane = tid & 31;
    const int lr   = lane & 15;
    const int lh   = lane >> 4;
    const int wm   = wid & 1;
    const int wn   = wid >> 1;
    const int Mb   = blockIdx.y * 128 + wm * 64;
    const int NbB  = blockIdx.x * 128;           // block N base (TDM tile)
    const unsigned lds_base = (unsigned)(size_t)(void*)&bufB[0][0];

    // --- TDM descriptor (CDNA5 ISA): 3-D tile,
    //     x = 8 dwords (j), y = 128 columns (n), z = 2 K-groups (g).
    auto issue_tile = [&](int k, int bufsel) {
        unsigned long long gaddr = (unsigned long long)(const unsigned char*)Xp
            + ((((unsigned long long)b * (unsigned)G + (unsigned)(k >> 4)) * NPOS
                + (unsigned)NbB) * 8ull) * 4ull;
        tdm_g0_t g0 = { 1u,                                   // count=1, user desc
                        lds_base + (unsigned)bufsel * 8192u,  // lds_addr (bytes)
                        (unsigned)gaddr,                      // global_addr[31:0]
                        (unsigned)((gaddr >> 32) & 0x01FFFFFFull) | 0x80000000u };
                                                              // addr[56:32] | type=2
        tdm_g1_t g1 = { (int)(2u << 16),          // wg_mask=0, data_size=4B
                        (int)(8u << 16),          // tensor_dim0 = 8
                        (int)(9216u << 16),       // tensor_dim1 = 9216
                        (int)(8u << 16),          // tile_dim0 = 8
                        (int)(128u | (2u << 16)), // tile_dim1=128, tile_dim2=2
                        (int)8,                   // tensor_dim0_stride = 8
                        (int)(0x2000u << 16),     // tensor_dim1_stride lo16 (73728)
                        (int)1 };                 // tensor_dim1_stride hi
        tdm_g23_t g2 = { 2, 0, 0, 0 };            // tensor_dim2 = 2
        tdm_g23_t g3 = { 0, 0, 0, 0 };
        tdm_g1_t  g4 = { 0, 0, 0, 0, 0, 0, 0, 0 }; // extended words (unused)
        __builtin_amdgcn_tensor_load_to_lds(g0, g1, g2, g3, g4, 0);
    };

    f32x8 acc[4][2];
#pragma unroll
    for (int mt = 0; mt < 4; ++mt)
#pragma unroll
        for (int nt = 0; nt < 2; ++nt)
#pragma unroll
            for (int r = 0; r < 8; ++r) acc[mt][nt][r] = 0.f;

    if (wid == 0) {
        issue_tile(0, 0);
        __builtin_amdgcn_s_wait_tensorcnt(0);
    }
    __syncthreads();

    for (int k = 0, it = 0; k < Kpad; k += 32, ++it) {
        const int cur = it & 1;
        if (wid == 0 && (k + 32) < Kpad) issue_tile(k + 32, cur ^ 1);

        BfFrag a[4], bf[2];
#pragma unroll
        for (int mt = 0; mt < 4; ++mt) {
            const int row = Mb + mt * 16 + lr;
            const u32x4* pa = (const u32x4*)(Wp + (size_t)row * Kpad + k + lh * 16);
            a[mt].q[0] = pa[0];          // K = k + 16*lh      (V0..V3)
            a[mt].q[1] = pa[1];          // K = k + 16*lh + 8  (V4..V7)
        }
#pragma unroll
        for (int nt = 0; nt < 2; ++nt) {
            const u32x4* pb =
                (const u32x4*)&bufB[cur][(unsigned)((lh * 128) + wn * 32 + nt * 16 + lr) * 8u];
            bf[nt].q[0] = pb[0];
            bf[nt].q[1] = pb[1];
        }
#pragma unroll
        for (int mt = 0; mt < 4; ++mt)
#pragma unroll
            for (int nt = 0; nt < 2; ++nt)
                acc[mt][nt] = __builtin_amdgcn_wmma_f32_16x16x32_bf16(
                    false, a[mt].v, false, bf[nt].v,
                    (short)0, acc[mt][nt], false, false);

        if (wid == 0 && (k + 32) < Kpad) __builtin_amdgcn_s_wait_tensorcnt(0);
        __syncthreads();
    }

#pragma unroll
    for (int mt = 0; mt < 4; ++mt)
#pragma unroll
        for (int nt = 0; nt < 2; ++nt)
#pragma unroll
            for (int r = 0; r < 8; ++r) {
                const int row = Mb + mt * 16 + r + 8 * lh;
                const int col = NbB + wn * 32 + nt * 16 + lr;
                const size_t idx = ((size_t)b * M + row) * NPOS + col;
                float v = acc[mt][nt][r];
                if (bias)  v += bias[row];
                if (resid) v += resid[idx];
                C[idx] = v;
            }
}

// ---------------------------------------------------------------------------
// K3: L2 row norms over spatial axis for q and k (blockIdx.y selects tensor)
// ---------------------------------------------------------------------------
__global__ __launch_bounds__(256) void k_rownorm(
    const float* __restrict__ q, const float* __restrict__ kv,
    float* __restrict__ nq, float* __restrict__ nk)
{
    __shared__ float red[256];
    const int row = blockIdx.x;           // b*256 + c
    const int b = row >> 8, c = row & 255;
    const int tid = threadIdx.x;
    const float* src = blockIdx.y ? (kv + ((size_t)b * 512 + c) * NPOS)
                                  : (q  + (size_t)row * NPOS);
    float s = 0.f;
    for (int i = tid; i < NPOS; i += 256) { float v = src[i]; s += v * v; }
    red[tid] = s; __syncthreads();
    for (int o = 128; o > 0; o >>= 1) {
        if (tid < o) red[tid] += red[tid + o];
        __syncthreads();
    }
    if (tid == 0) {
        float nm = fmaxf(sqrtf(red[0]), 1e-12f);
        (blockIdx.y ? nk : nq)[row] = nm;
    }
}

// ---------------------------------------------------------------------------
// K4: channel-attention logits via fp32 WMMA: logits[bh] = q_h * k_h^T
// ---------------------------------------------------------------------------
__global__ __launch_bounds__(128) void k_logits(
    const float* __restrict__ q, const float* __restrict__ kv,
    float* __restrict__ logits)           // [64][32][32]
{
    const int bh = blockIdx.x;
    const int b = bh >> 3, h = bh & 7;
    const int wid = threadIdx.x >> 5, lane = threadIdx.x & 31;
    const int lr = lane & 15, lh = lane >> 4;
    const int mi = wid >> 1, ni = wid & 1;

    const float* qrow = q  + ((size_t)b * 256 + h * 32 + mi * 16 + lr) * NPOS;
    const float* krow = kv + ((size_t)b * 512 + h * 32 + ni * 16 + lr) * NPOS;

    f32x8 acc;
#pragma unroll
    for (int r = 0; r < 8; ++r) acc[r] = 0.f;

    for (int n = 0; n < NPOS; n += 4) {
        f32x2 av = *(const f32x2*)(qrow + n + 2 * lh);
        f32x2 bv = *(const f32x2*)(krow + n + 2 * lh);
        acc = __builtin_amdgcn_wmma_f32_16x16x4_f32(
            false, av, false, bv, (short)0, acc, false, false);
    }
#pragma unroll
    for (int r = 0; r < 8; ++r) {
        const int cc = mi * 16 + r + 8 * lh;
        const int dd = ni * 16 + lr;
        logits[((size_t)bh * 32 + cc) * 32 + dd] = acc[r];
    }
}

// ---------------------------------------------------------------------------
// K5: fold q/k norms + tau, softmax over d
// ---------------------------------------------------------------------------
__global__ __launch_bounds__(32) void k_softmax(
    const float* __restrict__ logits, const float* __restrict__ nq,
    const float* __restrict__ nk, const float* __restrict__ log_tau,
    float* __restrict__ attn)
{
    const int bh = blockIdx.x, b = bh >> 3, h = bh & 7;
    const int c = threadIdx.x;
    const float tau = expf(log_tau[h]) + 1e-6f;
    const float inq = 1.f / nq[b * 256 + h * 32 + c];

    float row[32], mx = -1e30f;
#pragma unroll
    for (int d = 0; d < 32; ++d) {
        float l = logits[((size_t)bh * 32 + c) * 32 + d]
                  * inq / nk[b * 256 + h * 32 + d] * tau;
        row[d] = l; mx = fmaxf(mx, l);
    }
    float s = 0.f;
#pragma unroll
    for (int d = 0; d < 32; ++d) { row[d] = expf(row[d] - mx); s += row[d]; }
    const float inv = 1.f / s;
#pragma unroll
    for (int d = 0; d < 32; ++d)
        attn[((size_t)bh * 32 + c) * 32 + d] = row[d] * inv;
}

// ---------------------------------------------------------------------------
// K6: out_pre = attn @ v (K=32, per head), emitted in bf16 packed layout
// ---------------------------------------------------------------------------
__global__ __launch_bounds__(128) void k_attnv_pack(
    const float* __restrict__ attn, const float* __restrict__ kv,
    unsigned* __restrict__ dst)           // [B][GP2][NPOS][8] dwords
{
    __shared__ float sa[32 * 32];
    const int b = blockIdx.z, h = blockIdx.y;
    const int n = blockIdx.x * 128 + threadIdx.x;
    const int bh = b * 8 + h;

    for (int i = threadIdx.x; i < 1024; i += 128)
        sa[i] = attn[(size_t)bh * 1024 + i];
    __syncthreads();

    float vr[32];
    const float* vb = kv + ((size_t)b * 512 + 256 + h * 32) * NPOS + n;
#pragma unroll
    for (int d = 0; d < 32; ++d) vr[d] = vb[(size_t)d * NPOS];

#pragma unroll
    for (int half = 0; half < 2; ++half) {
        unsigned pk[8];
#pragma unroll
        for (int j = 0; j < 8; ++j) {
            const int ci = half * 16 + 2 * j;
            float o0 = 0.f, o1 = 0.f;
#pragma unroll
            for (int d = 0; d < 32; ++d) {
                o0 += sa[ci * 32 + d] * vr[d];
                o1 += sa[(ci + 1) * 32 + d] * vr[d];
            }
            pk[j] = (unsigned)f2bf(o0) | ((unsigned)f2bf(o1) << 16);
        }
        const int g = h * 2 + half;
        u32x4* pd = (u32x4*)(dst + (((size_t)b * GP2 + g) * NPOS + n) * 8);
        u32x4 lo = { pk[0], pk[1], pk[2], pk[3] };
        u32x4 hi = { pk[4], pk[5], pk[6], pk[7] };
        pd[0] = lo; pd[1] = hi;
    }
}

// ---------------------------------------------------------------------------
// Host-side launch orchestration
// ---------------------------------------------------------------------------
extern "C" void kernel_launch(void* const* d_in, const int* in_sizes, int n_in,
                              void* d_out, int out_size, void* d_ws, size_t ws_size,
                              hipStream_t stream)
{
    (void)in_sizes; (void)n_in; (void)out_size; (void)ws_size;

    const float* query    = (const float*)d_in[0];
    const float* context  = (const float*)d_in[1];
    const float* ln_q_w   = (const float*)d_in[2];
    const float* ln_q_b   = (const float*)d_in[3];
    const float* ln_kv_w  = (const float*)d_in[4];
    const float* ln_kv_b  = (const float*)d_in[5];
    const float* dw_q_w   = (const float*)d_in[6];
    const float* dw_kv_w  = (const float*)d_in[7];
    const float* q_proj_w = (const float*)d_in[8];
    const float* kv_proj_w= (const float*)d_in[9];
    const float* out_proj_w = (const float*)d_in[10];
    const float* out_proj_b = (const float*)d_in[11];
    const float* log_tau  = (const float*)d_in[12];
    float* out = (float*)d_out;

    char* ws = (char*)d_ws;
    size_t off = 0;
    auto take = [&](size_t bytes) -> char* {
        char* p = ws + off;
        off = (off + bytes + 255) & ~(size_t)255;
        return p;
    };

    unsigned short* wq_p  = (unsigned short*)take((size_t)256 * KPAD1 * 2);
    unsigned short* wkv_p = (unsigned short*)take((size_t)512 * KPAD1 * 2);
    unsigned short* wo_p  = (unsigned short*)take((size_t)256 * 256 * 2);
    unsigned* qd_p  = (unsigned*)take((size_t)8 * GP1 * NPOS * 32);
    unsigned* kd_p  = (unsigned*)take((size_t)8 * GP1 * NPOS * 32);
    float* qbuf  = (float*)take((size_t)8 * 256 * NPOS * 4);
    float* kvbuf = (float*)take((size_t)8 * 512 * NPOS * 4);
    float* nq    = (float*)take((size_t)8 * 256 * 4);
    float* nk    = (float*)take((size_t)8 * 256 * 4);
    float* logits= (float*)take((size_t)64 * 32 * 32 * 4);
    float* attn  = (float*)take((size_t)64 * 32 * 32 * 4);
    unsigned* op_p = (unsigned*)take((size_t)8 * GP2 * NPOS * 32);

    // K0: pack weights to bf16 (K-padded)
    k_pack_w<<<(256 * KPAD1 + 255) / 256, 256, 0, stream>>>(q_proj_w,  wq_p,  256, CINCH, KPAD1);
    k_pack_w<<<(512 * KPAD1 + 255) / 256, 256, 0, stream>>>(kv_proj_w, wkv_p, 512, CINCH, KPAD1);
    k_pack_w<<<(256 * 256  + 255) / 256, 256, 0, stream>>>(out_proj_w, wo_p,  256, 256,   256);

    // K1: coords + LN + depthwise conv + bf16 pack
    dim3 g1(HW / 16, HW / 16, 8);
    k_ln_dwconv_pack<<<g1, 256, 0, stream>>>(query,   ln_q_w,  ln_q_b,  dw_q_w,  qd_p);
    k_ln_dwconv_pack<<<g1, 256, 0, stream>>>(context, ln_kv_w, ln_kv_b, dw_kv_w, kd_p);

    // K2: q / kv projections (bf16 WMMA + TDM staging)
    k_gemm_bf16<<<dim3(NPOS / 128, 256 / 128, 8), 256, 0, stream>>>(
        wq_p,  qd_p, qbuf,  256, KPAD1, nullptr, nullptr);
    k_gemm_bf16<<<dim3(NPOS / 128, 512 / 128, 8), 256, 0, stream>>>(
        wkv_p, kd_p, kvbuf, 512, KPAD1, nullptr, nullptr);

    // K3: spatial L2 norms for q and k
    k_rownorm<<<dim3(8 * 256, 2), 256, 0, stream>>>(qbuf, kvbuf, nq, nk);

    // K4: channel-attention logits (fp32 WMMA, K = 9216)
    k_logits<<<64, 128, 0, stream>>>(qbuf, kvbuf, logits);

    // K5: scale by 1/(|q||k|) * tau and softmax
    k_softmax<<<64, 32, 0, stream>>>(logits, nq, nk, log_tau, attn);

    // K6: attn @ v, packed bf16 for final GEMM
    k_attnv_pack<<<dim3(NPOS / 128, 8, 8), 128, 0, stream>>>(attn, kvbuf, op_p);

    // K7: out_proj + bias + residual (bf16 WMMA GEMM with epilogue)
    k_gemm_bf16<<<dim3(NPOS / 128, 256 / 128, 8), 256, 0, stream>>>(
        wo_p, op_p, out, 256, 256, out_proj_b, query);
}
